// RLRNN_42013370089776
// MI455X (gfx1250) — matrix operand order repaired and checked by reading, not compiled, not run on hardware
//
#include <hip/hip_runtime.h>

// ---------------------------------------------------------------------------
// RL-RNN forward for MI455X (gfx1250, wave32, WMMA bf16 16x16x32)
// B=64, T=2048, H=256, NI=4, NA=2, TAU=2 (leak=0.5), CLAMP=1000, EPS=1e-4
// Scan kernel: persistent single-WGP, Whh resident in LDS, h state in VGPRs,
// u_t double-buffered in LDS via GLOBAL_LOAD_ASYNC_TO_LDS_B128 (ASYNCcnt).
// ---------------------------------------------------------------------------

typedef __attribute__((ext_vector_type(16))) __bf16 v16bf;
typedef __attribute__((ext_vector_type(8)))  float  v8f;

#define B_   64
#define T_   2048
#define H_   256

// ---------------- WMMA fragment loaders (contiguous-k layouts, wave32) -------

// A fragment: 16x32 bf16 tile, row-major source, row stride ldk (elements).
// lane L: row = L%16 ; elem j = K (L/16)*16 + j  (16 contiguous).
__device__ __forceinline__ v16bf load_a_frag(const __bf16* base, int ldk, int lane) {
    const int m  = lane & 15;
    const int k0 = (lane >> 4) * 16;
    const __bf16* p = base + m * ldk + k0;
    v16bf a;
#pragma unroll
    for (int i = 0; i < 8; ++i) a[i] = p[i];
#pragma unroll
    for (int i = 0; i < 8; ++i) a[8 + i] = p[8 + i];
    return a;
}

// B fragment for X @ W^T: B[k][n] = W[n][k]. W row-major, row stride ldk.
// lane L: col n = L%16 ; elem j = K (L/16)*16 + j  (16 contiguous of W row n).
__device__ __forceinline__ v16bf load_b_fragT(const __bf16* w, int ldk, int lane) {
    const int n  = lane & 15;
    const int k0 = (lane >> 4) * 16;
    const __bf16* p = w + n * ldk + k0;
    v16bf b;
#pragma unroll
    for (int i = 0; i < 16; ++i) b[i] = p[i];
    return b;
}

// ---------------- K0: weight prep (Whh diag fix + bf16 casts) --------------
__global__ void prep_weights_kernel(const float* __restrict__ Whh_raw,
                                    const float* __restrict__ Wih,
                                    const float* __restrict__ W2,
                                    __bf16* whh_b, __bf16* wih_b, __bf16* w2_b) {
    int i = blockIdx.x * 256 + threadIdx.x;           // 3 * 65536 total
    if (i < H_ * H_) {
        int r = i >> 8, c = i & 255;
        float v = (r == c) ? -0.5f : Whh_raw[i];      // -1/TAU on diagonal
        whh_b[i] = (__bf16)v;
    } else if (i < 2 * H_ * H_) {
        int j = i - H_ * H_;
        wih_b[j] = (__bf16)Wih[j];
    } else if (i < 3 * H_ * H_) {
        int j = i - 2 * H_ * H_;
        w2_b[j] = (__bf16)W2[j];
    }
}

// ---------------- K1: x = relu(concat(reward,state,last) @ W1^T + b1) ------
// Output row-major (T*B, H) as bf16, row = t*B + b.
__global__ void prep_x_kernel(const float* __restrict__ state,
                              const float* __restrict__ last_action,
                              const float* __restrict__ reward,
                              const float* __restrict__ W1,
                              const float* __restrict__ b1,
                              __bf16* __restrict__ X) {
    long idx = (long)blockIdx.x * 256 + threadIdx.x;  // T*B*H
    int hcol  = (int)(idx & 255);
    long row  = idx >> 8;                             // t*B + b
    int t = (int)(row >> 6);
    int b = (int)(row & 63);
    long base = (long)b * T_ + t;
    const float* w = W1 + hcol * 8;
    float s = b1[hcol];
    s += w[0] * reward[base * 2 + 0] + w[1] * reward[base * 2 + 1];
    s += w[2] * state[base * 4 + 0] + w[3] * state[base * 4 + 1];
    s += w[4] * state[base * 4 + 2] + w[5] * state[base * 4 + 3];
    s += w[6] * last_action[base * 2 + 0] + w[7] * last_action[base * 2 + 1];
    X[idx] = (__bf16)(s > 0.f ? s : 0.f);
}

// ---------------- K2: u = x @ Wih^T  (M=131072, N=256, K=256) --------------
__global__ void gemm_u_kernel(const __bf16* __restrict__ X,
                              const __bf16* __restrict__ WihB,
                              float* __restrict__ U) {
    int gwave = (blockIdx.x * blockDim.x + threadIdx.x) >> 5;
    int lane  = threadIdx.x & 31;
    int nt = gwave & 15;        // 16 n-tiles
    int mt = gwave >> 4;        // 8192 m-tiles
    const __bf16* Abase = X + (size_t)(mt * 16) * H_;
    const __bf16* Bbase = WihB + (size_t)(nt * 16) * H_;
    v8f acc = {};
#pragma unroll
    for (int kt = 0; kt < 8; ++kt) {
        v16bf a = load_a_frag(Abase + kt * 32, H_, lane);
        v16bf b = load_b_fragT(Bbase + kt * 32, H_, lane);
        acc = __builtin_amdgcn_wmma_f32_16x16x32_bf16(false, a, false, b,
                                                      (short)0, acc, false, false);
    }
    int n     = nt * 16 + (lane & 15);
    int mbase = mt * 16 + (lane >> 4) * 8;
#pragma unroll
    for (int r = 0; r < 8; ++r) U[(size_t)(mbase + r) * H_ + n] = acc[r];
}

// ---------------- K3: persistent single-WGP recurrent scan -----------------
// LDS: Whh bf16 (128KB) + relu(h) bf16 (32KB) + u double-buffer f32 (128KB)
//    = 288KB < 320KB.  h state lives in VGPRs (16 f32/lane, D-layout owned).
// Per step: 32 waves x 2 tiles x 8 wmma_bf16; u[t+1] prefetched with
// global_load_async_to_lds_b128 while step t computes.
__device__ __forceinline__ void async_copy16(const float* g, float* l) {
    uint32_t loff  = (uint32_t)(uintptr_t)l;     // LDS flat addr truncates to DS offset
    uint64_t gaddr = (uint64_t)(uintptr_t)g;
    asm volatile("global_load_async_to_lds_b128 %0, %1, off"
                 :: "v"(loff), "v"(gaddr) : "memory");
}

__global__ __launch_bounds__(1024) void scan_kernel(const float* __restrict__ hidden_in,
                                                    const float* __restrict__ U,
                                                    const __bf16* __restrict__ WhhB,
                                                    __bf16* __restrict__ YS,
                                                    float* __restrict__ hidden_out) {
    extern __shared__ char smem[];
    __bf16* whh = (__bf16*)smem;                      // 256*256 bf16 = 131072 B
    __bf16* rh  = (__bf16*)(smem + 131072);           // 64*256 bf16  =  32768 B
    float*  ub  = (float*)(smem + 163840);            // 2 * 64*256 f32 = 131072 B

    const int tid  = threadIdx.x;
    const int lane = tid & 31;
    const int wave = tid >> 5;   // 0..31

    for (int i = tid; i < H_ * H_; i += 1024) whh[i] = WhhB[i];

    // per-lane owned (m,n) positions (D-fragment layout) + h in registers
    float hreg[2][8];
    int mB[2], nC[2];
#pragma unroll
    for (int tt = 0; tt < 2; ++tt) {
        int tile = wave * 2 + tt;       // 0..63
        int mt = tile >> 4, nt = tile & 15;
        mB[tt] = mt * 16 + (lane >> 4) * 8;
        nC[tt] = nt * 16 + (lane & 15);
#pragma unroll
        for (int r = 0; r < 8; ++r)
            hreg[tt][r] = hidden_in[(mB[tt] + r) * H_ + nC[tt]];
    }

    // prefetch u[0] into buffer 0: 64KB = 1024 threads x 4 x b128
    {
        const float* g = U + tid * 16;
        float* l = ub + tid * 16;
#pragma unroll
        for (int j = 0; j < 4; ++j) async_copy16(g + j * 4, l + j * 4);
    }

    for (int t = 0; t < T_; ++t) {
        // phase 1: publish relu(h) bf16 from registers (owner lanes)
#pragma unroll
        for (int tt = 0; tt < 2; ++tt)
#pragma unroll
            for (int r = 0; r < 8; ++r) {
                float v = hreg[tt][r];
                rh[(mB[tt] + r) * H_ + nC[tt]] = (__bf16)(v > 0.f ? v : 0.f);
            }

        // prefetch u[t+1] into the other buffer, then wait for u[t]'s 4 ops
        if (t + 1 < T_) {
            const float* g = U + (size_t)(t + 1) * B_ * H_ + tid * 16;
            float* l = ub + ((t + 1) & 1) * B_ * H_ + tid * 16;
#pragma unroll
            for (int j = 0; j < 4; ++j) async_copy16(g + j * 4, l + j * 4);
            asm volatile("s_wait_asynccnt 4" ::: "memory");   // prev step's 4 done
        } else {
            asm volatile("s_wait_asynccnt 0" ::: "memory");
        }
        __syncthreads();   // rh + u[t] LDS visible to all waves

        const float* Ut = ub + (t & 1) * B_ * H_;
#pragma unroll
        for (int tt = 0; tt < 2; ++tt) {
            int tile = wave * 2 + tt;
            int mt = tile >> 4, nt = tile & 15;
            v8f acc = {};
            const __bf16* Abase = rh + mt * 16 * H_;
            const __bf16* Bbase = whh + nt * 16 * H_;
#pragma unroll
            for (int kt = 0; kt < 8; ++kt) {
                v16bf a = load_a_frag(Abase + kt * 32, H_, lane);
                v16bf b = load_b_fragT(Bbase + kt * 32, H_, lane);
                acc = __builtin_amdgcn_wmma_f32_16x16x32_bf16(false, a, false, b,
                                                              (short)0, acc, false, false);
            }
#pragma unroll
            for (int r = 0; r < 8; ++r) {
                int m = mB[tt] + r, n = nC[tt];
                float hn = 0.5f * hreg[tt][r] + Ut[m * H_ + n] + acc[r];
                hn = fminf(fmaxf(hn, -1000.f), 1000.f);
                hreg[tt][r] = hn;
                YS[((size_t)t * B_ + m) * H_ + n] = (__bf16)hn;
            }
        }
        __syncthreads();   // all reads of rh / u[t] done before next overwrite
    }

#pragma unroll
    for (int tt = 0; tt < 2; ++tt)
#pragma unroll
        for (int r = 0; r < 8; ++r)
            hidden_out[(mB[tt] + r) * H_ + nC[tt]] = hreg[tt][r];
}

// ---------------- K4: z = relu(ys@W2^T + b2) ; y = z@W3^T + b3 -------------
__global__ void gemm_z_kernel(const __bf16* __restrict__ YS,
                              const __bf16* __restrict__ W2B,
                              const float* __restrict__ b2,
                              const float* __restrict__ W3,
                              const float* __restrict__ b3,
                              float* __restrict__ Y) {
    __shared__ __bf16 zlds[128 * H_];     // 64 KB
    int lane = threadIdx.x & 31;
    int wave = threadIdx.x >> 5;          // 0..7
    int mt = blockIdx.x * 8 + wave;       // global m-tile
    const __bf16* Abase = YS + (size_t)(mt * 16) * H_;
    for (int ntile = 0; ntile < 16; ++ntile) {
        v8f acc = {};
        const __bf16* Bbase = W2B + (size_t)(ntile * 16) * H_;
#pragma unroll
        for (int kt = 0; kt < 8; ++kt) {
            v16bf a = load_a_frag(Abase + kt * 32, H_, lane);
            v16bf b = load_b_fragT(Bbase + kt * 32, H_, lane);
            acc = __builtin_amdgcn_wmma_f32_16x16x32_bf16(false, a, false, b,
                                                          (short)0, acc, false, false);
        }
        int n    = ntile * 16 + (lane & 15);
        int mloc = wave * 16 + (lane >> 4) * 8;
        float bias = b2[n];
#pragma unroll
        for (int r = 0; r < 8; ++r) {
            float v = acc[r] + bias;
            zlds[(mloc + r) * H_ + n] = (__bf16)(v > 0.f ? v : 0.f);
        }
    }
    __syncthreads();
    // 256 threads -> 128 rows x NA=2
    int r = threadIdx.x >> 1;
    int a = threadIdx.x & 1;
    float s = b3[a];
    const float* w3 = W3 + a * H_;
    for (int k = 0; k < H_; ++k) s += (float)zlds[r * H_ + k] * w3[k];
    int grow = blockIdx.x * 128 + r;
    Y[(size_t)grow * 2 + a] = s;
}

// ---------------- K5: softmax / log-mix / exp ------------------------------
__global__ void finalize_kernel(const float* __restrict__ Y,
                                const float* __restrict__ Qs,
                                const float* __restrict__ reward,
                                const float* __restrict__ mix_w,
                                float* __restrict__ out) {
    int idx = blockIdx.x * 256 + threadIdx.x;   // 0 .. B*T-1
    int b = idx >> 11;                          // / T_
    int t = idx & (T_ - 1);
    size_t row = (size_t)t * B_ + b;            // y is (T,B,2)
    size_t bt  = (size_t)b * T_ + t;            // outputs are (B,T,2)

    float p0 = Y[row * 2 + 0], p1 = Y[row * 2 + 1];
    float mx = fmaxf(p0, p1);
    float e0 = __expf(p0 - mx), e1 = __expf(p1 - mx);
    float inv = 1.f / (e0 + e1);
    float z0 = (p0 == 0.f) ? 1e-4f : 0.f;
    float z1 = (p1 == 0.f) ? 1e-4f : 0.f;
    float lp0 = __logf(e0 * inv + z0), lp1 = __logf(e1 * inv + z1);

    float q0 = Qs[bt * 2 + 0], q1 = Qs[bt * 2 + 1];
    float qm = fmaxf(q0, q1);
    float f0 = __expf(q0 - qm), f1 = __expf(q1 - qm);
    float fin = 1.f / (f0 + f1);
    float zq0 = (q0 == 0.f) ? 1e-4f : 0.f;
    float zq1 = (q1 == 0.f) ? 1e-4f : 0.f;
    float lq0 = __logf(f0 * fin + zq0), lq1 = __logf(f1 * fin + zq1);

    int ri = (int)reward[bt * 2 + 0];           // 0 or 1
    float w0 = fminf(fmaxf(mix_w[ri * 2 + 0], -1.f), 1.f);
    float w1 = fminf(fmaxf(mix_w[ri * 2 + 1], -1.f), 1.f);
    float den = fabsf(w0) + fabsf(w1);
    float s0 = w0 / den, s1 = w1 / den;

    out[bt * 2 + 0] = __expf(lp0 * s0 + lq0 * s1);
    out[bt * 2 + 1] = __expf(lp1 * s0 + lq1 * s1);
}

// ---------------------------------------------------------------------------
extern "C" void kernel_launch(void* const* d_in, const int* in_sizes, int n_in,
                              void* d_out, int out_size, void* d_ws, size_t ws_size,
                              hipStream_t stream) {
    const float* state       = (const float*)d_in[0];
    const float* last_action = (const float*)d_in[1];
    const float* reward      = (const float*)d_in[2];
    const float* hidden_in   = (const float*)d_in[3];
    const float* W1          = (const float*)d_in[4];
    const float* b1          = (const float*)d_in[5];
    const float* Wih         = (const float*)d_in[6];
    const float* Whh_raw     = (const float*)d_in[7];
    const float* W2          = (const float*)d_in[8];
    const float* b2          = (const float*)d_in[9];
    const float* W3          = (const float*)d_in[10];
    const float* b3          = (const float*)d_in[11];
    const float* Qs          = (const float*)d_in[12];
    const float* mix_w       = (const float*)d_in[13];

    // workspace layout (bytes)
    char* ws = (char*)d_ws;
    const size_t NROW = (size_t)T_ * B_;               // 131072
    const size_t OFF_X   = 0;                          // bf16 NROW*H
    const size_t OFF_U   = OFF_X   + NROW * H_ * 2;    // f32  NROW*H
    const size_t OFF_YS  = OFF_U   + NROW * H_ * 4;    // bf16 NROW*H
    const size_t OFF_Y   = OFF_YS  + NROW * H_ * 2;    // f32  NROW*2
    const size_t OFF_WHH = OFF_Y   + NROW * 2 * 4;     // bf16 H*H
    const size_t OFF_WIH = OFF_WHH + (size_t)H_ * H_ * 2;
    const size_t OFF_W2  = OFF_WIH + (size_t)H_ * H_ * 2;

    __bf16* Xb   = (__bf16*)(ws + OFF_X);
    float*  U    = (float*)(ws + OFF_U);
    __bf16* YS   = (__bf16*)(ws + OFF_YS);
    float*  Y    = (float*)(ws + OFF_Y);
    __bf16* WhhB = (__bf16*)(ws + OFF_WHH);
    __bf16* WihB = (__bf16*)(ws + OFF_WIH);
    __bf16* W2B  = (__bf16*)(ws + OFF_W2);

    float* out_probs  = (float*)d_out;                 // (B,T,2) = 262144
    float* hidden_out = out_probs + (size_t)B_ * T_ * 2;

    // K0: weight prep (3*65536 elements)
    prep_weights_kernel<<<(3 * H_ * H_) / 256, 256, 0, stream>>>(
        Whh_raw, Wih, W2, WhhB, WihB, W2B);

    // K1: input layer (T*B*H elements)
    prep_x_kernel<<<(int)(NROW * H_ / 256), 256, 0, stream>>>(
        state, last_action, reward, W1, b1, Xb);

    // K2: u = x @ Wih^T   (131072 tiles, 8 waves/block)
    gemm_u_kernel<<<(int)(NROW / 16 * (H_ / 16) / 8), 256, 0, stream>>>(Xb, WihB, U);

    // K3: persistent scan, one WGP, 288KB dynamic LDS (Whh + rh + u dbl-buf)
    scan_kernel<<<1, 1024, 131072 + 32768 + 131072, stream>>>(
        hidden_in, U, WhhB, YS, hidden_out);

    // K4: output head (128 rows/block)
    gemm_z_kernel<<<(int)(NROW / 128), 256, 0, stream>>>(YS, W2B, b2, W3, b3, Y);

    // K5: elementwise finalize (B*T threads)
    finalize_kernel<<<(int)((size_t)B_ * T_ / 256), 256, 0, stream>>>(
        Y, Qs, reward, mix_w, out_probs);
}
